// MeshGraphNet_26474178413320
// MI455X (gfx1250) — matrix operand order repaired and checked by reading, not compiled, not run on hardware
//
#include <hip/hip_runtime.h>
#include <hip/hip_bf16.h>

#define HDIM   128
#define LDA    136        // padded bf16 LDS row stride (272B -> 4-bank stagger per row)
#define TILE_M 128
#define NWAVE  8
#define EPSL   1e-5f
#define NBLK   15

typedef __attribute__((ext_vector_type(16))) __bf16 v16bf;
typedef __attribute__((ext_vector_type(8)))  __bf16 v8bf;
typedef __attribute__((ext_vector_type(4)))  __bf16 v4bf;
typedef __attribute__((ext_vector_type(8)))  float  v8f;

// ---------------------------------------------------------------------------
// Convert + transpose fp32 weight [K, 128] -> bf16 [128, K] (n-major)
// ---------------------------------------------------------------------------
__global__ void prep_weight_kernel(const float* __restrict__ W,
                                   __bf16* __restrict__ Wt, int K) {
  int i = blockIdx.x * 256 + threadIdx.x;
  if (i < K * HDIM) {
    int k = i / HDIM, n = i % HDIM;
    Wt[(size_t)n * K + k] = (__bf16)W[(size_t)k * HDIM + n];
  }
}

// ---------------------------------------------------------------------------
// Generic WMMA MLP layer:
//   C[M,128] = epilogue( sum_t gather(A_t, idx_t)[M,128] @ W_t[128,128] + bias )
// epilogue: ReLU, or LayerNorm(gamma,beta) (+ optional residual).
// 256 threads = 8 waves; wave w computes rows 16w..16w+15 x all 128 cols
// using v_wmma_f32_16x16x32_bf16 with f32 accumulation.
// ---------------------------------------------------------------------------
__global__ __launch_bounds__(256)
void gemm_mlp_kernel(const float* __restrict__ A0, const int* __restrict__ idx0,
                     const float* __restrict__ A1, const int* __restrict__ idx1,
                     const float* __restrict__ A2, const int* __restrict__ idx2,
                     int nterms, int K,
                     const __bf16* __restrict__ Wt,   // [128][K] bf16, n-major
                     const float* __restrict__ bias,
                     const float* __restrict__ gamma, const float* __restrict__ beta,
                     const float* __restrict__ resid,
                     float* __restrict__ C, int M, int relu_out)
{
  __shared__ __align__(16) unsigned char lds_raw[TILE_M * LDA * sizeof(__bf16)]; // 34816 B
  __bf16* ldsA = (__bf16*)lds_raw;
  float*  ldsF = (float*)lds_raw;   // reused as 64 x 132 f32 epilogue scratch

  const int tid  = threadIdx.x;
  const int lane = tid & 31;
  const int wv   = tid >> 5;
  const int lw   = lane & 15;
  const int hi   = lane >> 4;      // 0: lanes 0-15, 1: lanes 16-31
  const int row0 = blockIdx.x * TILE_M;

  v8f acc[8];
  #pragma unroll
  for (int nt = 0; nt < 8; ++nt) acc[nt] = {};

  for (int t = 0; t < nterms; ++t) {
    const float* A   = (t == 0) ? A0 : ((t == 1) ? A1 : A2);
    const int*   idx = (t == 0) ? idx0 : ((t == 1) ? idx1 : idx2);
    __syncthreads();
    // Stage 128 gathered rows, fp32 -> bf16, into padded LDS.
    // One wave iteration = one full row (32 lanes x float4).
    for (int r = wv; r < TILE_M; r += NWAVE) {
      int gr = row0 + r;
      if (gr >= M) gr = M - 1;                       // uniform clamp (EXEC stays full)
      int srow = idx ? idx[gr] : gr;
      float4 f = *(const float4*)(A + (size_t)srow * HDIM + lane * 4);
      v4bf h;
      h[0] = (__bf16)f.x; h[1] = (__bf16)f.y; h[2] = (__bf16)f.z; h[3] = (__bf16)f.w;
      *(v4bf*)(ldsA + r * LDA + lane * 4) = h;
    }
    __syncthreads();

    const __bf16* wt = Wt + t * HDIM;   // term t occupies k = t*128 .. t*128+127
    #pragma unroll
    for (int kk = 0; kk < 4; ++kk) {
      // A fragment (16x32 bf16): lane lw = row, 16 contiguous K starting at hi*16
      const __bf16* ap = ldsA + (wv * 16 + lw) * LDA + kk * 32 + hi * 16;
      v8bf alo = *(const v8bf*)ap;         // ds_load_b128
      v8bf ahi = *(const v8bf*)(ap + 8);   // ds_load_b128
      v16bf a;
      #pragma unroll
      for (int j = 0; j < 8; ++j) { a[j] = alo[j]; a[8 + j] = ahi[j]; }

      #pragma unroll
      for (int nt = 0; nt < 8; ++nt) {
        // B fragment (32x16 bf16): lane col n, 16 contiguous K starting at hi*16
        int n = nt * 16 + lw;
        v16bf b = *(const v16bf*)(wt + (size_t)n * K + kk * 32 + hi * 16);
        acc[nt] = __builtin_amdgcn_wmma_f32_16x16x32_bf16(
            false, a, false, b, (short)0, acc[nt], false, false);
      }
    }
  }

  // ---- epilogue: two 64-row passes through LDS (fits in the 34KB A buffer) ----
  for (int pass = 0; pass < 2; ++pass) {
    __syncthreads();
    if ((wv >> 2) == pass) {
      int rloc = (wv & 3) * 16;
      #pragma unroll
      for (int nt = 0; nt < 8; ++nt) {
        int col = nt * 16 + lw;
        #pragma unroll
        for (int v = 0; v < 8; ++v)
          ldsF[(rloc + v + 8 * hi) * 132 + col] = acc[nt][v];   // C layout: M = v + 8*hi
      }
    }
    __syncthreads();
    if (tid < 64) {
      int gr = row0 + pass * 64 + tid;
      if (gr < M) {
        const float* x = ldsF + tid * 132;
        float* out = C + (size_t)gr * HDIM;
        if (gamma) {
          float s = 0.f, s2 = 0.f;
          #pragma unroll 4
          for (int c = 0; c < HDIM; ++c) {
            float v = x[c] + bias[c];
            s += v; s2 += v * v;
          }
          float mu  = s  * (1.0f / HDIM);
          float var = s2 * (1.0f / HDIM) - mu * mu;
          float rs  = rsqrtf(var + EPSL);
          const float* rr = resid ? (resid + (size_t)gr * HDIM) : nullptr;
          for (int c = 0; c < HDIM; c += 4) {
            float4 o;
            o.x = (x[c+0] + bias[c+0] - mu) * rs * gamma[c+0] + beta[c+0];
            o.y = (x[c+1] + bias[c+1] - mu) * rs * gamma[c+1] + beta[c+1];
            o.z = (x[c+2] + bias[c+2] - mu) * rs * gamma[c+2] + beta[c+2];
            o.w = (x[c+3] + bias[c+3] - mu) * rs * gamma[c+3] + beta[c+3];
            if (rr) { o.x += rr[c+0]; o.y += rr[c+1]; o.z += rr[c+2]; o.w += rr[c+3]; }
            *(float4*)(out + c) = o;
          }
        } else {
          for (int c = 0; c < HDIM; c += 4) {
            float4 o;
            o.x = x[c+0] + bias[c+0]; o.y = x[c+1] + bias[c+1];
            o.z = x[c+2] + bias[c+2]; o.w = x[c+3] + bias[c+3];
            if (relu_out) {
              o.x = fmaxf(o.x, 0.f); o.y = fmaxf(o.y, 0.f);
              o.z = fmaxf(o.z, 0.f); o.w = fmaxf(o.w, 0.f);
            }
            *(float4*)(out + c) = o;
          }
        }
      }
    }
  }
}

// ---------------------------------------------------------------------------
// Tiny-K encoder first layer: Y = relu(X[M,K] @ W[K,128] + b), K in {3,4}
// ---------------------------------------------------------------------------
__global__ void enc_first_kernel(const float* __restrict__ X, int K,
                                 const float* __restrict__ W, const float* __restrict__ b,
                                 float* __restrict__ Y, int M) {
  int gid = blockIdx.x * 256 + threadIdx.x;
  int r = gid >> 7, c = gid & 127;
  if (r >= M) return;
  float acc = b[c];
  for (int k = 0; k < K; ++k) acc += X[(size_t)r * K + k] * W[k * HDIM + c];
  Y[(size_t)r * HDIM + c] = fmaxf(acc, 0.f);
}

// Decoder head: Y[M,2] = X[M,128] @ W[128,2] + b
__global__ void dec_last_kernel(const float* __restrict__ X,
                                const float* __restrict__ W, const float* __restrict__ b,
                                float* __restrict__ Y, int M) {
  int gid = blockIdx.x * 256 + threadIdx.x;
  int r = gid >> 1, o = gid & 1;
  if (r >= M) return;
  float acc = b[o];
  for (int k = 0; k < HDIM; ++k) acc += X[(size_t)r * HDIM + k] * W[k * 2 + o];
  Y[(size_t)r * 2 + o] = acc;
}

__global__ void zero_kernel(float* __restrict__ p, long n) {
  long i = (long)blockIdx.x * 256 + threadIdx.x;
  long stride = (long)gridDim.x * 256;
  for (; i < n; i += stride) p[i] = 0.f;
}

// agg[dst[e]] += ef[e] (segment_sum), 4 columns per thread
__global__ void scatter_add_kernel(const float* __restrict__ ef, const int* __restrict__ dst,
                                   float* __restrict__ agg, int E) {
  long gid = (long)blockIdx.x * 256 + threadIdx.x;
  int e  = (int)(gid >> 5);
  int cg = (int)(gid & 31) * 4;
  if (e >= E) return;
  int d = dst[e];
  const float* s = ef + (size_t)e * HDIM + cg;
  float* a = agg + (size_t)d * HDIM + cg;
  atomicAdd(a + 0, s[0]); atomicAdd(a + 1, s[1]);
  atomicAdd(a + 2, s[2]); atomicAdd(a + 3, s[3]);
}

// ---------------------------------------------------------------------------
extern "C" void kernel_launch(void* const* d_in, const int* in_sizes, int n_in,
                              void* d_out, int out_size, void* d_ws, size_t ws_size,
                              hipStream_t stream) {
  (void)n_in; (void)out_size; (void)ws_size;
  int p = 0;
  const float* node_features = (const float*)d_in[p++];
  const float* edge_features = (const float*)d_in[p++];
  const int N = in_sizes[0] / 4;
  const int E = in_sizes[1] / 3;

  struct MLP { const float *W0,*b0,*W1,*b1,*W2,*b2; };
  struct LNP { const float *g,*b; };
  auto takeMLP = [&](MLP& m) {
    m.W0=(const float*)d_in[p++]; m.b0=(const float*)d_in[p++];
    m.W1=(const float*)d_in[p++]; m.b1=(const float*)d_in[p++];
    m.W2=(const float*)d_in[p++]; m.b2=(const float*)d_in[p++];
  };
  auto takeLN = [&](LNP& l) {
    l.g=(const float*)d_in[p++]; l.b=(const float*)d_in[p++];
  };

  MLP ne; LNP ne_ln; takeMLP(ne); takeLN(ne_ln);
  MLP ee; LNP ee_ln; takeMLP(ee); takeLN(ee_ln);
  MLP eb[NBLK]; LNP eb_ln[NBLK];
  for (int i = 0; i < NBLK; ++i) { takeMLP(eb[i]); takeLN(eb_ln[i]); }
  MLP nb[NBLK]; LNP nb_ln[NBLK];
  for (int i = 0; i < NBLK; ++i) { takeMLP(nb[i]); takeLN(nb_ln[i]); }
  MLP de; takeMLP(de);
  const int* src = (const int*)d_in[p++];
  const int* dst = (const int*)d_in[p++];

  // ---- workspace carve ----
  char* ws = (char*)d_ws;
  size_t off = 0;
  auto carve = [&](size_t bytes) -> char* {
    char* r = ws + off; off += (bytes + 255) & ~(size_t)255; return r;
  };
  float* nf  = (float*)carve((size_t)N * HDIM * 4);
  float* ef  = (float*)carve((size_t)E * HDIM * 4);
  float* t1  = (float*)carve((size_t)E * HDIM * 4);   // shared temps (E >= N)
  float* t2  = (float*)carve((size_t)E * HDIM * 4);
  float* agg = (float*)carve((size_t)N * HDIM * 4);

  // ---- pre-transpose/convert all WMMA weights to bf16 [128][K] ----
  auto carveW = [&](const float* W, int K) -> __bf16* {
    __bf16* wt = (__bf16*)carve((size_t)K * HDIM * 2);
    int total = K * HDIM;
    prep_weight_kernel<<<(total + 255) / 256, 256, 0, stream>>>(W, wt, K);
    return wt;
  };
  __bf16 *ne_W1t = carveW(ne.W1, 128), *ne_W2t = carveW(ne.W2, 128);
  __bf16 *ee_W1t = carveW(ee.W1, 128), *ee_W2t = carveW(ee.W2, 128);
  __bf16 *eb_W0t[NBLK], *eb_W1t[NBLK], *eb_W2t[NBLK];
  __bf16 *nb_W0t[NBLK], *nb_W1t[NBLK], *nb_W2t[NBLK];
  for (int i = 0; i < NBLK; ++i) {
    eb_W0t[i] = carveW(eb[i].W0, 384);
    eb_W1t[i] = carveW(eb[i].W1, 128);
    eb_W2t[i] = carveW(eb[i].W2, 128);
    nb_W0t[i] = carveW(nb[i].W0, 256);
    nb_W1t[i] = carveW(nb[i].W1, 128);
    nb_W2t[i] = carveW(nb[i].W2, 128);
  }
  __bf16 *de_W0t = carveW(de.W0, 128), *de_W1t = carveW(de.W1, 128);

  auto gemm = [&](const float* A0, const int* i0, const float* A1, const int* i1,
                  const float* A2, const int* i2, int nterms, int K,
                  const __bf16* Wt, const float* b, const float* g, const float* bt,
                  const float* res, float* out, int M, int relu) {
    gemm_mlp_kernel<<<(M + TILE_M - 1) / TILE_M, 256, 0, stream>>>(
        A0, i0, A1, i1, A2, i2, nterms, K, Wt, b, g, bt, res, out, M, relu);
  };

  // ---- node encoder ----
  enc_first_kernel<<<((size_t)N * 128 + 255) / 256, 256, 0, stream>>>(
      node_features, 4, ne.W0, ne.b0, t1, N);
  gemm(t1, nullptr, nullptr, nullptr, nullptr, nullptr, 1, 128, ne_W1t, ne.b1,
       nullptr, nullptr, nullptr, t2, N, 1);
  gemm(t2, nullptr, nullptr, nullptr, nullptr, nullptr, 1, 128, ne_W2t, ne.b2,
       ne_ln.g, ne_ln.b, nullptr, nf, N, 0);

  // ---- edge encoder ----
  enc_first_kernel<<<((size_t)E * 128 + 255) / 256, 256, 0, stream>>>(
      edge_features, 3, ee.W0, ee.b0, t1, E);
  gemm(t1, nullptr, nullptr, nullptr, nullptr, nullptr, 1, 128, ee_W1t, ee.b1,
       nullptr, nullptr, nullptr, t2, E, 1);
  gemm(t2, nullptr, nullptr, nullptr, nullptr, nullptr, 1, 128, ee_W2t, ee.b2,
       ee_ln.g, ee_ln.b, nullptr, ef, E, 0);

  // ---- processor blocks ----
  long aggN = (long)N * HDIM;
  for (int i = 0; i < NBLK; ++i) {
    // edge MLP: cat([nf[src], nf[dst], ef]) -> LN -> +ef
    gemm(nf, src, nf, dst, ef, nullptr, 3, 384, eb_W0t[i], eb[i].b0,
         nullptr, nullptr, nullptr, t1, E, 1);
    gemm(t1, nullptr, nullptr, nullptr, nullptr, nullptr, 1, 128, eb_W1t[i], eb[i].b1,
         nullptr, nullptr, nullptr, t2, E, 1);
    gemm(t2, nullptr, nullptr, nullptr, nullptr, nullptr, 1, 128, eb_W2t[i], eb[i].b2,
         eb_ln[i].g, eb_ln[i].b, ef, ef, E, 0);
    // aggregate: agg = segment_sum(ef, dst)
    zero_kernel<<<4096, 256, 0, stream>>>(agg, aggN);
    scatter_add_kernel<<<((long)E * 32 + 255) / 256, 256, 0, stream>>>(ef, dst, agg, E);
    // node MLP: cat([agg, nf]) -> LN -> +nf
    gemm(agg, nullptr, nf, nullptr, nullptr, nullptr, 2, 256, nb_W0t[i], nb[i].b0,
         nullptr, nullptr, nullptr, t1, N, 1);
    gemm(t1, nullptr, nullptr, nullptr, nullptr, nullptr, 1, 128, nb_W1t[i], nb[i].b1,
         nullptr, nullptr, nullptr, t2, N, 1);
    gemm(t2, nullptr, nullptr, nullptr, nullptr, nullptr, 1, 128, nb_W2t[i], nb[i].b2,
         nb_ln[i].g, nb_ln[i].b, nf, nf, N, 0);
  }

  // ---- decoder ----
  gemm(nf, nullptr, nullptr, nullptr, nullptr, nullptr, 1, 128, de_W0t, de.b0,
       nullptr, nullptr, nullptr, t1, N, 1);
  gemm(t1, nullptr, nullptr, nullptr, nullptr, nullptr, 1, 128, de_W1t, de.b1,
       nullptr, nullptr, nullptr, t2, N, 1);
  dec_last_kernel<<<((size_t)N * 2 + 255) / 256, 256, 0, stream>>>(
      t2, de.W2, de.b2, (float*)d_out, N);
}
